// GemmaForCausalLM_85822036509136
// MI455X (gfx1250) — compile-verified, matrix operand order checked by reading, not run on hardware
//
#include <hip/hip_runtime.h>
#include <hip/hip_bf16.h>
#include <math.h>

// ---------------------------------------------------------------------------
// Gemma-2-style 2-layer forward for MI455X (gfx1250), wave32 + WMMA bf16.
// GEMMs: double-buffered LDS, register-staged global loads, optional TDM
// (tensor_load_to_lds) for the bf16 A operand.
// ---------------------------------------------------------------------------

typedef __attribute__((ext_vector_type(8)))  __bf16 v8bf;
typedef __attribute__((ext_vector_type(16))) __bf16 v16bf;
typedef __attribute__((ext_vector_type(8)))  float  v8f;

#define S_LEN  1024
#define HDIM   2048
#define NHEADS 8
#define HEADD  256
#define IDIM   8192
#define VOCAB  32768
#define QKVN   2560     // (NH + 2*NKV) * HD
#define WIN_   512
#define LDSA   32       // A tile row stride (bf16) -> 64B rows, TDM-friendly
#define LDSB   48       // B tile row stride (bf16) -> 96B rows, 16B aligned frags

#if defined(__has_builtin)
#if __has_builtin(__builtin_amdgcn_tensor_load_to_lds) && \
    __has_builtin(__builtin_amdgcn_s_wait_tensorcnt)
#define USE_TDM 1
#endif
#endif
#ifndef USE_TDM
#define USE_TDM 0
#endif

#if USE_TDM
typedef __attribute__((ext_vector_type(4))) unsigned int u32x4;
typedef __attribute__((ext_vector_type(4))) int          i32x4;
typedef __attribute__((ext_vector_type(8))) int          i32x8;

// Issue one TDM 2D-tile load: 32 (K) x 128 (M) bf16 tile, row stride K elems,
// from global (tile origin gsrc) into LDS at byte offset lds_off (unpadded,
// 64B rows). D# layout per CDNA5 ISA 8.3/8.4. One DMA per block (wave 0).
static __device__ __forceinline__ void tdm_load_a_tile(const __bf16* gsrc,
                                                       unsigned lds_off, int K) {
  const unsigned long long ga = (unsigned long long)(uintptr_t)gsrc;
  u32x4 g0;
  g0[0] = 1u;                                           // count=1, user mode
  g0[1] = lds_off;                                      // LDS byte address
  g0[2] = (unsigned)ga;                                 // global_addr[31:0]
  g0[3] = (unsigned)((ga >> 32) & 0x01ffffffull)        // global_addr[56:32]
          | (2u << 30);                                 // type = 2 (image)
  i32x8 g1;
  g1[0] = 1 << 16;                                      // data_size=1 -> 2 bytes
  g1[1] = (K & 0xffff) << 16;                           // tensor_dim0[15:0]
  g1[2] = ((unsigned)K >> 16) | (S_LEN << 16);          // dim0[31:16] | dim1[15:0]
  g1[3] = (32 << 16);                                   // dim1[31:16]=0 | tile_dim0=32
  g1[4] = 128;                                          // tile_dim1=128, tile_dim2=0
  g1[5] = K;                                            // tensor_dim0_stride[31:0]
  g1[6] = 0;
  g1[7] = 0;
  i32x4 z4 = {0, 0, 0, 0};
#if __clang_major__ >= 23
  i32x8 z8 = {0, 0, 0, 0, 0, 0, 0, 0};
  __builtin_amdgcn_tensor_load_to_lds(g0, g1, z4, z4, z8, 0);
#else
  __builtin_amdgcn_tensor_load_to_lds(g0, g1, z4, z4, 0);
#endif
}
#endif  // USE_TDM

// ---------------------------------------------------------------------------
// LDS fragment loader following the CDNA5 16-bit A/B VGPR layout:
// lane L holds row (L&15); K-base = (L>>4)*8; 8 elems at +0 and 8 at +16.
// ---------------------------------------------------------------------------
static __device__ __forceinline__ v16bf load_frag_lds(const __bf16* base,
                                                      int stride, int row, int kb) {
  const __bf16* p = base + row * stride + kb;
  v8bf lo = *(const v8bf*)(p);
  v8bf hi = *(const v8bf*)(p + 16);
  v16bf f;
#pragma unroll
  for (int i = 0; i < 8; ++i) { f[i] = lo[i]; f[i + 8] = hi[i]; }
  return f;
}

// ---------------------------------------------------------------------------
// C[M,N] = A[M,K](bf16) x W[N,K](f32, converted to bf16 in LDS)^T
// mode 0: plain f32 store; mode 1: tanh(x/cap)*cap (logit soft-cap)
// Block tile 128x128, K-step 32, 8 waves, 2x4 WMMA tiles per wave,
// double-buffered LDS, one barrier per K-step, loads overlapped with WMMA.
// Requires M%128==0, N%128==0, K%32==0 (true for all call sites).
// ---------------------------------------------------------------------------
__global__ __launch_bounds__(256) void gemm_bf16_wmma(
    const __bf16* __restrict__ A, const float* __restrict__ W,
    float* __restrict__ C, int M, int N, int K, int mode, float cap)
{
  __shared__ __bf16 As[2][128 * LDSA];   // 2 x 8 KB
  __shared__ __bf16 Bs[2][128 * LDSB];   // 2 x 12 KB

  const int tid  = threadIdx.x;
  const int lane = tid & 31;
  const int wave = tid >> 5;
  const int wm   = wave >> 1;            // 0..3 : 32-row stripe
  const int wn   = wave & 1;             // 0..1 : 64-col stripe
  const int mbase = blockIdx.y * 128;
  const int nbase = blockIdx.x * 128;

  v8f acc[2][4];
#pragma unroll
  for (int t = 0; t < 2; ++t)
#pragma unroll
    for (int u = 0; u < 4; ++u)
#pragma unroll
      for (int g = 0; g < 8; ++g) acc[t][u][g] = 0.0f;

  // per-thread staging coordinates
  int brow[4], bcol[4];
#pragma unroll
  for (int i = 0; i < 4; ++i) {
    const int c = tid + (i << 8);
    brow[i] = c >> 3; bcol[i] = (c & 7) << 2;      // B: 4 float4 per thread
  }
#if !USE_TDM
  int arow[2], acol[2];
#pragma unroll
  for (int i = 0; i < 2; ++i) {
    const int c = tid + (i << 8);
    arow[i] = c >> 2; acol[i] = (c & 3) << 3;      // A: 2 x v8bf per thread
  }
  v8bf areg[2];
#endif
  float4 breg[4];

  // ---- prologue: issue tile 0 loads ----
#if USE_TDM
  if (wave == 0)
    tdm_load_a_tile(A + (size_t)mbase * K,
                    (unsigned)(uintptr_t)&As[0][0], K);
#else
#pragma unroll
  for (int i = 0; i < 2; ++i)
    areg[i] = *(const v8bf*)&A[(size_t)(mbase + arow[i]) * K + acol[i]];
#endif
#pragma unroll
  for (int i = 0; i < 4; ++i)
    breg[i] = *(const float4*)&W[(size_t)(nbase + brow[i]) * K + bcol[i]];

  const int ktiles = K >> 5;
  const int lrow = lane & 15;
  const int kb   = (lane >> 4) << 3;

  for (int kt = 0; kt < ktiles; ++kt) {
    const int cur = kt & 1;
    const int nxt = cur ^ 1;

    // ---- finish staging tile kt into LDS[cur] ----
#pragma unroll
    for (int i = 0; i < 4; ++i) {
      __bf16* dst = &Bs[cur][brow[i] * LDSB + bcol[i]];
      dst[0] = (__bf16)breg[i].x; dst[1] = (__bf16)breg[i].y;
      dst[2] = (__bf16)breg[i].z; dst[3] = (__bf16)breg[i].w;
    }
#if USE_TDM
    if (wave == 0) __builtin_amdgcn_s_wait_tensorcnt(0);
#else
#pragma unroll
    for (int i = 0; i < 2; ++i)
      *(v8bf*)&As[cur][arow[i] * LDSA + acol[i]] = areg[i];
#endif
    __syncthreads();

    // ---- issue tile kt+1 loads (overlap with WMMA below) ----
    if (kt + 1 < ktiles) {
      const int k0n = (kt + 1) << 5;
#if USE_TDM
      if (wave == 0)
        tdm_load_a_tile(A + (size_t)mbase * K + k0n,
                        (unsigned)(uintptr_t)&As[nxt][0], K);
#else
#pragma unroll
      for (int i = 0; i < 2; ++i)
        areg[i] = *(const v8bf*)&A[(size_t)(mbase + arow[i]) * K + k0n + acol[i]];
#endif
#pragma unroll
      for (int i = 0; i < 4; ++i)
        breg[i] = *(const float4*)&W[(size_t)(nbase + brow[i]) * K + k0n + bcol[i]];
    }

    // ---- compute on LDS[cur] ----
    v16bf af[2], bfv[4];
#pragma unroll
    for (int t = 0; t < 2; ++t)
      af[t] = load_frag_lds(&As[cur][0], LDSA, wm * 32 + t * 16 + lrow, kb);
#pragma unroll
    for (int u = 0; u < 4; ++u)
      bfv[u] = load_frag_lds(&Bs[cur][0], LDSB, wn * 64 + u * 16 + lrow, kb);

#pragma unroll
    for (int t = 0; t < 2; ++t)
#pragma unroll
      for (int u = 0; u < 4; ++u)
        acc[t][u] = __builtin_amdgcn_wmma_f32_16x16x32_bf16(
            false, af[t], false, bfv[u], (short)0, acc[t][u], false, false);
  }

  // ---- store: C/D layout: VGPR g, lanes 0-15 -> M=g, lanes 16-31 -> M=g+8 ----
  const int lhalf = lane >> 4, lcol = lane & 15;
#pragma unroll
  for (int t = 0; t < 2; ++t)
#pragma unroll
    for (int u = 0; u < 4; ++u)
#pragma unroll
      for (int g = 0; g < 8; ++g) {
        const int row = mbase + wm * 32 + t * 16 + g + lhalf * 8;
        const int col = nbase + wn * 64 + u * 16 + lcol;
        float v = acc[t][u][g];
        if (mode == 1) v = tanhf(v * (1.0f / cap)) * cap;
        C[(size_t)row * N + col] = v;
      }
}

// ---------------------------------------------------------------------------
// h[s] = embed_w[token[s]] * sqrt(H)
// ---------------------------------------------------------------------------
__global__ __launch_bounds__(256) void embed_kernel(
    const int* __restrict__ tok, const float* __restrict__ E, float* __restrict__ h)
{
  const int s = blockIdx.x;
  const int t = tok[s];
  const float scale = 45.25483399593904f;  // sqrt(2048)
  for (int i = threadIdx.x; i < HDIM; i += 256)
    h[(size_t)s * HDIM + i] = E[(size_t)t * HDIM + i] * scale;
}

// ---------------------------------------------------------------------------
// out(bf16) = rmsnorm(x, w)   (row length 2048)
// ---------------------------------------------------------------------------
__global__ __launch_bounds__(256) void rmsnorm_to_bf16(
    const float* __restrict__ x, const float* __restrict__ w,
    __bf16* __restrict__ out)
{
  __shared__ float red[256];
  const int row = blockIdx.x, tid = threadIdx.x;
  const float* xr = x + (size_t)row * HDIM;
  float ss = 0.f;
  for (int i = tid; i < HDIM; i += 256) { const float v = xr[i]; ss += v * v; }
  red[tid] = ss; __syncthreads();
  for (int t = 128; t > 0; t >>= 1) { if (tid < t) red[tid] += red[tid + t]; __syncthreads(); }
  const float inv = rsqrtf(red[0] * (1.0f / HDIM) + 1e-6f);
  for (int i = tid; i < HDIM; i += 256)
    out[(size_t)row * HDIM + i] = (__bf16)(xr[i] * inv * (1.0f + w[i]));
}

// ---------------------------------------------------------------------------
// h[s] += rmsnorm(x[s], w)    (post-attn / post-ffw residual)
// ---------------------------------------------------------------------------
__global__ __launch_bounds__(256) void rmsnorm_add(
    const float* __restrict__ x, const float* __restrict__ w, float* __restrict__ h)
{
  __shared__ float red[256];
  const int row = blockIdx.x, tid = threadIdx.x;
  const float* xr = x + (size_t)row * HDIM;
  float ss = 0.f;
  for (int i = tid; i < HDIM; i += 256) { const float v = xr[i]; ss += v * v; }
  red[tid] = ss; __syncthreads();
  for (int t = 128; t > 0; t >>= 1) { if (tid < t) red[tid] += red[tid + t]; __syncthreads(); }
  const float inv = rsqrtf(red[0] * (1.0f / HDIM) + 1e-6f);
  for (int i = tid; i < HDIM; i += 256)
    h[(size_t)row * HDIM + i] += xr[i] * inv * (1.0f + w[i]);
}

// ---------------------------------------------------------------------------
// Split qkv row, apply RoPE to q (+1/16 scaling) and k, copy v.
// ---------------------------------------------------------------------------
__global__ __launch_bounds__(256) void rope_kernel(
    const float* __restrict__ qkv, const int* __restrict__ pos,
    float* __restrict__ Q, float* __restrict__ K, float* __restrict__ V)
{
  const int s = blockIdx.x, tid = threadIdx.x;
  const float p = (float)pos[s];
  const float lnth = 9.210340371976184f;  // ln(10000)
  const float* row = qkv + (size_t)s * QKVN;

  for (int idx = tid; idx < NHEADS * 128; idx += 256) {
    const int hh = idx >> 7, i = idx & 127;
    const float ang = p * __expf(-((float)i) * (lnth / 128.0f));
    const float c = cosf(ang), sn = sinf(ang);
    const float x1 = row[hh * HEADD + i];
    const float x2 = row[hh * HEADD + 128 + i];
    float* q = Q + ((size_t)s * NHEADS + hh) * HEADD;
    q[i]       = (x1 * c - x2 * sn) * 0.0625f;   // QPA^-0.5 = 256^-0.5
    q[128 + i] = (x1 * sn + x2 * c) * 0.0625f;
  }
  if (tid < 128) {
    const int i = tid;
    const float ang = p * __expf(-((float)i) * (lnth / 128.0f));
    const float c = cosf(ang), sn = sinf(ang);
    const float x1 = row[HDIM + i];
    const float x2 = row[HDIM + 128 + i];
    K[(size_t)s * HEADD + i]       = x1 * c - x2 * sn;
    K[(size_t)s * HEADD + 128 + i] = x1 * sn + x2 * c;
  }
  V[(size_t)s * HEADD + tid] = row[HDIM + HEADD + tid];
}

// ---------------------------------------------------------------------------
// One block per (query s, head). tanh soft-cap, causal (+ local band), softmax,
// prob @ V; output bf16 for the o-proj WMMA GEMM.
// ---------------------------------------------------------------------------
__global__ __launch_bounds__(256) void attention_kernel(
    const float* __restrict__ Q, const float* __restrict__ K,
    const float* __restrict__ V, __bf16* __restrict__ O, int local)
{
  __shared__ float qs[HEADD];
  __shared__ float sc[S_LEN];
  __shared__ float red[256];
  const int s = blockIdx.x, hh = blockIdx.y, tid = threadIdx.x;

  qs[tid] = Q[((size_t)s * NHEADS + hh) * HEADD + tid];
  __syncthreads();

  const int lo = local ? ((s - (WIN_ - 1)) > 0 ? (s - (WIN_ - 1)) : 0) : 0;
  const int n  = s - lo + 1;

  for (int i = tid; i < n; i += 256) {
    const float* kr = K + (size_t)(lo + i) * HEADD;
    float dot = 0.f;
#pragma unroll 8
    for (int d = 0; d < HEADD; ++d) dot = fmaf(qs[d], kr[d], dot);
    sc[i] = tanhf(dot * (1.0f / 50.0f)) * 50.0f;   // ATTN_CAP
  }
  __syncthreads();

  float m = -3.0e38f;
  for (int i = tid; i < n; i += 256) m = fmaxf(m, sc[i]);
  red[tid] = m; __syncthreads();
  for (int t = 128; t > 0; t >>= 1) { if (tid < t) red[tid] = fmaxf(red[tid], red[tid + t]); __syncthreads(); }
  m = red[0]; __syncthreads();

  float ssum = 0.f;
  for (int i = tid; i < n; i += 256) { const float e = __expf(sc[i] - m); sc[i] = e; ssum += e; }
  red[tid] = ssum; __syncthreads();
  for (int t = 128; t > 0; t >>= 1) { if (tid < t) red[tid] += red[tid + t]; __syncthreads(); }
  const float inv = 1.0f / red[0];
  __syncthreads();

  float acc = 0.f;
  for (int i = 0; i < n; ++i)
    acc = fmaf(sc[i], V[(size_t)(lo + i) * HEADD + tid], acc);
  O[(size_t)s * HDIM + hh * HEADD + tid] = (__bf16)(acc * inv);
}

// ---------------------------------------------------------------------------
// mlp(bf16) = gelu_tanh(gate) * up
// ---------------------------------------------------------------------------
__global__ __launch_bounds__(256) void gelu_mul_kernel(
    const float* __restrict__ g, const float* __restrict__ u,
    __bf16* __restrict__ o, int n)
{
  const int idx = blockIdx.x * blockDim.x + threadIdx.x;
  const int stride = gridDim.x * blockDim.x;
  for (int i = idx; i < n; i += stride) {
    const float x = g[i];
    const float t = 0.5f * x * (1.0f + tanhf(0.7978845608f * (x + 0.044715f * x * x * x)));
    o[i] = (__bf16)(t * u[i]);
  }
}

// ---------------------------------------------------------------------------
// Host orchestration
// ---------------------------------------------------------------------------
extern "C" void kernel_launch(void* const* d_in, const int* in_sizes, int n_in,
                              void* d_out, int out_size, void* d_ws, size_t ws_size,
                              hipStream_t stream) {
  (void)in_sizes; (void)n_in; (void)out_size; (void)ws_size;

  const int*   tok      = (const int*)d_in[0];
  const int*   pos      = (const int*)d_in[1];
  const float* embed_w  = (const float*)d_in[4];
  const float* final_ln = (const float*)d_in[27];

  // workspace layout (bytes)
  char* ws = (char*)d_ws;
  float*  h     = (float*) (ws + 0);                 //  8 MB  S*H f32
  __bf16* xb    = (__bf16*)(ws + 8388608);           //  4 MB  S*H bf16
  float*  qkv   = (float*) (ws + 12582912);          // 10 MB  S*2560 f32
  float*  qbuf  = (float*) (ws + 23068672);          //  8 MB  S*NH*HD f32
  float*  kbuf  = (float*) (ws + 31457280);          //  1 MB  S*HD f32
  float*  vbuf  = (float*) (ws + 32505856);          //  1 MB  S*HD f32
  __bf16* attnb = (__bf16*)(ws + 33554432);          //  4 MB  S*H bf16
  float*  tmp   = (float*) (ws + 37748736);          //  8 MB  S*H f32
  __bf16* mlpb  = (__bf16*)(ws + 46137344);          // 16 MB  S*I bf16
  // gate/up staged in d_out (128 MB f32) before logits are written
  float*  gate  = (float*)d_out;
  float*  up    = (float*)d_out + (size_t)S_LEN * IDIM;

  embed_kernel<<<S_LEN, 256, 0, stream>>>(tok, embed_w, h);

  for (int l = 0; l < 2; ++l) {
    const int b = l ? 16 : 5;
    const float* qkv_w  = (const float*)d_in[b + 0];
    const float* o_w    = (const float*)d_in[b + 1];
    const float* gate_w = (const float*)d_in[b + 2];
    const float* up_w   = (const float*)d_in[b + 3];
    const float* down_w = (const float*)d_in[b + 4];
    const float* ln_in  = (const float*)d_in[b + 5];
    const float* ln_pa  = (const float*)d_in[b + 6];
    const float* ln_pf  = (const float*)d_in[b + 7];
    const float* ln_pof = (const float*)d_in[b + 8];

    rmsnorm_to_bf16<<<S_LEN, 256, 0, stream>>>(h, ln_in, xb);
    gemm_bf16_wmma<<<dim3(QKVN / 128, S_LEN / 128), 256, 0, stream>>>(
        xb, qkv_w, qkv, S_LEN, QKVN, HDIM, 0, 0.f);
    rope_kernel<<<S_LEN, 256, 0, stream>>>(qkv, pos, qbuf, kbuf, vbuf);
    attention_kernel<<<dim3(S_LEN, NHEADS), 256, 0, stream>>>(
        qbuf, kbuf, vbuf, attnb, l == 0 ? 1 : 0);
    gemm_bf16_wmma<<<dim3(HDIM / 128, S_LEN / 128), 256, 0, stream>>>(
        attnb, o_w, tmp, S_LEN, HDIM, HDIM, 0, 0.f);
    rmsnorm_add<<<S_LEN, 256, 0, stream>>>(tmp, ln_pa, h);

    rmsnorm_to_bf16<<<S_LEN, 256, 0, stream>>>(h, ln_pf, xb);
    gemm_bf16_wmma<<<dim3(IDIM / 128, S_LEN / 128), 256, 0, stream>>>(
        xb, gate_w, gate, S_LEN, IDIM, HDIM, 0, 0.f);
    gemm_bf16_wmma<<<dim3(IDIM / 128, S_LEN / 128), 256, 0, stream>>>(
        xb, up_w, up, S_LEN, IDIM, HDIM, 0, 0.f);
    gelu_mul_kernel<<<8192, 256, 0, stream>>>(gate, up, mlpb, S_LEN * IDIM);
    gemm_bf16_wmma<<<dim3(HDIM / 128, S_LEN / 128), 256, 0, stream>>>(
        mlpb, down_w, tmp, S_LEN, HDIM, IDIM, 0, 0.f);
    rmsnorm_add<<<S_LEN, 256, 0, stream>>>(tmp, ln_pof, h);
  }

  rmsnorm_to_bf16<<<S_LEN, 256, 0, stream>>>(h, final_ln, xb);
  gemm_bf16_wmma<<<dim3(VOCAB / 128, S_LEN / 128), 256, 0, stream>>>(
      xb, embed_w, (float*)d_out, S_LEN, VOCAB, HDIM, 1, 30.0f);
}
